// MoEBlockBase_42752104465026
// MI455X (gfx1250) — matrix operand order repaired and checked by reading, not compiled, not run on hardware
//
#include <hip/hip_runtime.h>

typedef __attribute__((ext_vector_type(16))) __bf16 v16bf;
typedef __attribute__((ext_vector_type(8)))  float  v8f;
typedef __attribute__((ext_vector_type(4)))  unsigned int u32x4;
typedef __attribute__((ext_vector_type(8)))  int          i32x8;
typedef __attribute__((ext_vector_type(4)))  int          i32x4;

#define BB   16
#define CC   64
#define HH   128
#define WW_  128
#define EE   8
#define GHD  16
#define KTOT 576      /* C * 3 * 3 */
#define HWN  16384    /* H * W     */
#define ASTR 584      /* A row stride (LDS AND global weff): mult of 8, conflict-free */
#define XCP  72       /* XsT ci-dim padded stride (mult of 8) */
#define APANEL (CC * ASTR)   /* 37376 bf16 elements per (batch) A panel */

// ---------------------------------------------------------------------------
// 1) Global average pool: one block per (b,c), reduce 16384 floats.
// ---------------------------------------------------------------------------
__global__ __launch_bounds__(256) void gap_kernel(const float* __restrict__ x,
                                                  float* __restrict__ pooled) {
  __shared__ float red[256];
  const int bc = blockIdx.x;                     // b*64 + c
  const float* p = x + ((size_t)bc << 14);
  float s = 0.f;
  for (int i = threadIdx.x; i < HWN; i += 256) s += p[i];
  red[threadIdx.x] = s;
  __syncthreads();
  for (int off = 128; off > 0; off >>= 1) {
    if ((int)threadIdx.x < off) red[threadIdx.x] += red[threadIdx.x + off];
    __syncthreads();
  }
  if (threadIdx.x == 0) pooled[bc] = red[0] * (1.0f / HWN);
}

// ---------------------------------------------------------------------------
// 2) Gating MLP + softmax + top-2 sparse renormalized probs (tiny: 1 wave).
// ---------------------------------------------------------------------------
__global__ void gate_kernel(const float* __restrict__ pooled,
                            const float* __restrict__ wg1, const float* __restrict__ bg1,
                            const float* __restrict__ wg2, const float* __restrict__ bg2,
                            float* __restrict__ probs) {
  const int b = threadIdx.x;
  if (b >= BB) return;
  float hbuf[GHD];
  for (int g = 0; g < GHD; ++g) {
    float acc = bg1[g];
    for (int c = 0; c < CC; ++c) acc += pooled[b * CC + c] * wg1[c * GHD + g];
    hbuf[g] = acc > 0.f ? acc : 0.f;
  }
  float pr[EE];
  float mx = -1e30f;
  for (int e = 0; e < EE; ++e) {
    float acc = bg2[e];
    for (int g = 0; g < GHD; ++g) acc += hbuf[g] * wg2[g * EE + e];
    pr[e] = acc;                                  // TAU = 1
    mx = fmaxf(mx, acc);
  }
  float psum = 0.f;
  for (int e = 0; e < EE; ++e) { pr[e] = __expf(pr[e] - mx); psum += pr[e]; }
  for (int e = 0; e < EE; ++e) pr[e] /= psum;
  int i1 = 0;
  for (int e = 1; e < EE; ++e) if (pr[e] > pr[i1]) i1 = e;
  int i2 = -1; float m2 = -1.f;
  for (int e = 0; e < EE; ++e) if (e != i1 && pr[e] > m2) { m2 = pr[e]; i2 = e; }
  const float s = pr[i1] + m2 + 1e-8f;
  for (int e = 0; e < EE; ++e)
    probs[b * EE + e] = (e == i1 || e == i2) ? pr[e] / s : 0.f;
}

// ---------------------------------------------------------------------------
// 3a) Fold probs into per-batch bias.
// ---------------------------------------------------------------------------
__global__ void beff_kernel(const float* __restrict__ probs,
                            const float* __restrict__ bexp,
                            float* __restrict__ beff) {
  const int idx = blockIdx.x * 256 + threadIdx.x;    // b*64 + c
  if (idx >= BB * CC) return;
  const int b = idx >> 6, c = idx & 63;
  float acc = 0.f;
  for (int e = 0; e < EE; ++e) acc += probs[b * EE + e] * bexp[e * CC + c];
  beff[idx] = acc;
}

// ---------------------------------------------------------------------------
// 3b) Fold probs into per-batch bf16 weights, TAP-MAJOR K order and LDS-image
//     layout: weff[b][c*ASTR + k'],  k' = (kh*3+kw)*64 + ci.
//     The pad region [576..583] per row is never read; the TDM copies the
//     padded panel verbatim, so the LDS image keeps the conflict-free stride.
// ---------------------------------------------------------------------------
__global__ void weff_kernel(const float* __restrict__ probs,
                            const float* __restrict__ wexp,
                            __bf16* __restrict__ weff) {
  const int idx = blockIdx.x * 256 + threadIdx.x;    // over B*C*KTOT = 589824
  if (idx >= BB * CC * KTOT) return;
  const int b   = idx / (CC * KTOT);
  const int rem = idx - b * CC * KTOT;
  const int c   = rem / KTOT;
  const int kp  = rem - c * KTOT;                    // permuted k'
  const int tap = kp >> 6;                           // kh*3+kw
  const int ci  = kp & 63;
  float acc = 0.f;
  for (int e = 0; e < EE; ++e)
    acc += probs[b * EE + e] *
           wexp[(((size_t)e * CC + c) * CC + ci) * 9 + tap];
  weff[(size_t)b * APANEL + c * ASTR + kp] = (__bf16)acc;
}

// ---------------------------------------------------------------------------
// 4) Implicit-GEMM 3x3 conv, V_WMMA_F32_16X16X32_BF16.
//    Block = (batch b, one full output row h): D[64 x 128] += A[64x576]*B[576x128]
//    A panel arrives via the Tensor Data Mover (one flat DMA, overlapped with
//    the waves' x-halo transpose staging); the K-loop is pure ds_load_b128 +
//    wmma with no barriers and no global traffic.
// ---------------------------------------------------------------------------
__global__ __launch_bounds__(256) void moe_conv_kernel(
    const float*  __restrict__ x,
    const __bf16* __restrict__ weff,
    const float*  __restrict__ beff,
    float*        __restrict__ out) {
  __shared__ __attribute__((aligned(16))) __bf16 As[CC][ASTR];       // 74,752 B
  __shared__ __attribute__((aligned(16))) __bf16 XsT[3][130][XCP];   // 56,160 B

  const int b = blockIdx.x >> 7;       // 128 row-blocks per batch
  const int h = blockIdx.x & 127;

  const int tid   = threadIdx.x;
  const int lane  = tid & 31;
  const int wave  = tid >> 5;
  const int mtile = wave >> 1;         // 0..3  -> channels mtile*16..+15
  const int npair = wave & 1;          // 0..1  -> columns npair*64..+63

  // ---- A panel via TDM: wave 0 issues one flat 1-D tensor DMA ----
  if (wave == 0) {
    const unsigned long long ga =
        (unsigned long long)(uintptr_t)(weff + (size_t)b * APANEL);
    const unsigned int lds_addr = (unsigned int)(uintptr_t)&As[0][0];
    // D# group0: count=1 | lds_addr | global_addr[56:0] | type=2
    u32x4 g0;
    g0[0] = 1u;                                     // count=1, user mode
    g0[1] = lds_addr;                               // LDS byte address
    g0[2] = (unsigned int)ga;                       // global_addr[31:0] -> bits 95:64
    g0[3] = (unsigned int)((ga >> 32) & 0x01FFFFFFu) | 0x80000000u; // addr[56:32] | type=2
    // D# group1: data_size=2B; tensor_dim0 = tile_dim0 = APANEL; dim1 = 1
    i32x8 g1;
    g1[0] = (int)(1u << 16);                        // workgroup_mask=0, data_size=1 (2B)
    g1[1] = (int)((APANEL & 0xFFFFu) << 16);        // tensor_dim0[15:0]
    g1[2] = (int)(((APANEL >> 16) & 0xFFFFu) | (1u << 16)); // tensor_dim0[31:16], tensor_dim1=1
    g1[3] = (int)((APANEL & 0xFFFFu) << 16);        // tile_dim0 = APANEL
    g1[4] = 1;                                      // tile_dim1 = 1, tile_dim2 = 0
    g1[5] = (int)APANEL;                            // tensor_dim0_stride[31:0]
    g1[6] = 0;                                      // stride hi, dim1_stride lo
    g1[7] = 0;
    i32x4 gz4 = {0, 0, 0, 0};
    i32x8 gz8 = {0, 0, 0, 0, 0, 0, 0, 0};
    __builtin_amdgcn_tensor_load_to_lds(g0, g1, gz4, gz4, gz8, 0);
  }

  // ---- stage XsT[kh][j][ci] = x[b, ci, h+kh-1, j-1]  (zero-padded halo) ----
  for (int r = wave; r < 192; r += 8) {            // 192 = 3 kh * 64 ci rows
    const int kh  = r >> 6;
    const int ci  = r & 63;
    const int hin = h + kh - 1;
    const float* src = x + (((size_t)(b * CC + ci)) << 14) + (size_t)(hin > 0 ? hin : 0) * WW_;
    const bool hok = (unsigned)hin < (unsigned)HH;
    for (int j = lane; j < 130; j += 32) {
      const int w = j - 1;
      float v = 0.f;
      if (hok && (unsigned)w < (unsigned)WW_) v = src[w];
      XsT[kh][j][ci] = (__bf16)v;
    }
  }
  if (wave == 0) __builtin_amdgcn_s_wait_tensorcnt(0);   // A panel landed
  __syncthreads();

  // ---- K loop: LDS-only, barrier-free ----
  const int m     = lane & 15;
  const int halfk = lane >> 4;
  const int arow  = mtile * 16 + m;
  const int n     = lane & 15;

  v8f acc[4] = {{0.f,0.f,0.f,0.f,0.f,0.f,0.f,0.f},
                {0.f,0.f,0.f,0.f,0.f,0.f,0.f,0.f},
                {0.f,0.f,0.f,0.f,0.f,0.f,0.f,0.f},
                {0.f,0.f,0.f,0.f,0.f,0.f,0.f,0.f}};

#pragma unroll
  for (int k0 = 0; k0 < KTOT; k0 += 32) {
    const int tap = k0 >> 6;            // constant after unroll
    const int kh  = tap / 3;            // constants: no runtime division
    const int kw  = tap - kh * 3;
    const int ci0 = k0 & 63;            // 0 or 32

    // A fragment: K = k0 + halfk*16 + {0..15}
    v16bf a;
#pragma unroll
    for (int t = 0; t < 8; ++t) a[t]     = As[arow][k0 + halfk * 16 + t];
#pragma unroll
    for (int t = 0; t < 8; ++t) a[8 + t] = As[arow][k0 + halfk * 16 + 8 + t];

#pragma unroll
    for (int i = 0; i < 4; ++i) {
      const int col = npair * 64 + i * 16 + n;   // output column (w)
      const int j   = col + kw;                  // halo-adjusted x column
      // B fragment: K = halfk*16 + {0..15} -> ci contiguous: one 32B LDS run
      v16bf bf;
#pragma unroll
      for (int t = 0; t < 16; ++t)
        bf[t] = XsT[kh][j][ci0 + halfk * 16 + t];

      acc[i] = __builtin_amdgcn_wmma_f32_16x16x32_bf16(
          false, a, false, bf, (short)0, acc[i], false, false);
    }
  }

  // ---- epilogue: D(m,n) lane map: N = lane&15, M = r + 8*(lane>>4) ----
  const int nloc  = lane & 15;
  const int halfm = lane >> 4;
#pragma unroll
  for (int i = 0; i < 4; ++i) {
    const int ww   = npair * 64 + i * 16 + nloc;
    const int spat = h * WW_ + ww;
#pragma unroll
    for (int r = 0; r < 8; ++r) {
      const int c = mtile * 16 + halfm * 8 + r;
      const size_t off = (((size_t)(b * CC + c)) << 14) + spat;
      out[off] = acc[i][r] + beff[b * CC + c] + x[off];
    }
  }
}

// ---------------------------------------------------------------------------
// launch
// ---------------------------------------------------------------------------
extern "C" void kernel_launch(void* const* d_in, const int* in_sizes, int n_in,
                              void* d_out, int out_size, void* d_ws, size_t ws_size,
                              hipStream_t stream) {
  const float* x    = (const float*)d_in[0];
  const float* wg1  = (const float*)d_in[1];
  const float* bg1  = (const float*)d_in[2];
  const float* wg2  = (const float*)d_in[3];
  const float* bg2  = (const float*)d_in[4];
  const float* wexp = (const float*)d_in[5];
  const float* bexp = (const float*)d_in[6];
  float* out = (float*)d_out;

  // workspace layout
  float*  probs  = (float*)d_ws;                 // 16*8
  float*  pooled = probs + 128;                  // 16*64
  float*  beff   = pooled + 1024;                // 16*64
  __bf16* weff   = (__bf16*)(beff + 1024);       // 16 padded A panels (16B aligned)

  gap_kernel <<<BB * CC, 256, 0, stream>>>(x, pooled);
  gate_kernel<<<1, 32, 0, stream>>>(pooled, wg1, bg1, wg2, bg2, probs);
  beff_kernel<<<(BB * CC + 255) / 256, 256, 0, stream>>>(probs, bexp, beff);
  weff_kernel<<<(BB * CC * KTOT + 255) / 256, 256, 0, stream>>>(probs, wexp, weff);
  moe_conv_kernel<<<BB * HH, 256, 0, stream>>>(x, weff, beff, out);
}
